// PyG_PointNet2_Alpha_Predictor_29832842838025
// MI455X (gfx1250) — matrix operand order repaired and checked by reading, not compiled, not run on hardware
//
#include <hip/hip_runtime.h>

// ---------------------------------------------------------------------------
// PointNet++ alpha predictor, CDNA5 (gfx1250) WMMA implementation.
// All GEMMs run as f16 x f16 -> f32-accumulate V_WMMA_F32_16X16X32_F16.
// ---------------------------------------------------------------------------

typedef _Float16 h16;
typedef __attribute__((ext_vector_type(16))) _Float16 v16h;
typedef __attribute__((ext_vector_type(8)))  _Float16 v8h;
typedef __attribute__((ext_vector_type(8)))  float    v8f;

// ---------------------------------------------------------------------------
// Weight prep: w_f32[K][M] (jax x@w layout) -> Wt_f16[M][Kp], K zero-padded.
// Column-of-B = contiguous-K row of Wt so WMMA B fragments are b128 loads.
// ---------------------------------------------------------------------------
__global__ void __launch_bounds__(256) prep_weight(const float* __restrict__ w,
                                                   h16* __restrict__ wt,
                                                   int K, int M, int Kp) {
  int m = blockIdx.x;
  int k = blockIdx.y * 256 + threadIdx.x;
  if (k >= Kp) return;
  wt[(long)m * Kp + k] = (k < K) ? (h16)w[(long)k * M + m] : (h16)0.f;
}

// l23[i] = l2_idx[l3_idx[i]]  (double indirection used by sa3 / fp3 / segmax)
__global__ void __launch_bounds__(256) make_l23(const int* __restrict__ l3_idx,
                                                const int* __restrict__ l2_idx,
                                                int* __restrict__ l23, int n) {
  int i = blockIdx.x * 256 + threadIdx.x;
  if (i < n) l23[i] = l2_idx[l3_idx[i]];
}

// ---------------------------------------------------------------------------
// Generic staging: out_f16[N, Wp] = [ gather(srcA) | gather(srcB) | cast_f32(srcC) | 0-pad ]
// idx* == null -> identity row; idxA divided by divA (for batch = l23/N).
// ---------------------------------------------------------------------------
__global__ void __launch_bounds__(256) stage_concat(
    h16* __restrict__ out, int Wp,
    const h16* __restrict__ srcA, int wA, const int* __restrict__ idxA, int divA, int ldA,
    const h16* __restrict__ srcB, int wB, const int* __restrict__ idxB, int ldB,
    const float* __restrict__ srcC, int wC, const int* __restrict__ idxC, int ldC) {
  long row = blockIdx.x;
  int col = blockIdx.y * 256 + threadIdx.x;
  if (col >= Wp) return;
  h16 v = (h16)0.f;
  if (col < wA) {
    long r = idxA ? (long)(idxA[row] / divA) : row;
    v = srcA[r * ldA + col];
  } else if (col < wA + wB) {
    long r = idxB ? (long)idxB[row] : row;
    v = srcB[r * ldB + (col - wA)];
  } else if (col < wA + wB + wC) {
    long r = idxC ? (long)idxC[row] : row;
    v = (h16)srcC[r * ldC + (col - wA - wB)];
  }
  out[row * (long)Wp + col] = v;
}

// ---------------------------------------------------------------------------
// WMMA GEMM + bias + optional ReLU.
//   C_f16[N, M] = act( A_f16[N, Kp] @ Wt_f16[M, Kp]^T + bias_f32[M] )
// Block: 256 threads = 8 waves; wave computes 16 rows x 64 cols (4 wmma tiles,
// shared A fragment). Grid: (N/128, M/64). No divergence: EXEC stays all-1s
// as WMMA requires. Weights are L2-resident; A is streamed (prefetched).
// ---------------------------------------------------------------------------
__global__ void __launch_bounds__(256) gemm_bias_act(
    const h16* __restrict__ A, const h16* __restrict__ Wt,
    const float* __restrict__ bias, h16* __restrict__ C,
    int Kp, int M, int relu) {
  const int lane = threadIdx.x & 31;
  const int wave = threadIdx.x >> 5;
  const int m16  = lane & 15;   // A: row-in-tile / B,D: col-in-tile
  const int hi   = lane >> 4;
  const long row0 = (long)blockIdx.x * 128 + wave * 16;
  const int  col0 = blockIdx.y * 64;

  const h16* arow = A + (row0 + m16) * (long)Kp;
  const h16* bp0 = Wt + ((long)(col0 +  0 + m16)) * Kp + 16 * hi;
  const h16* bp1 = Wt + ((long)(col0 + 16 + m16)) * Kp + 16 * hi;
  const h16* bp2 = Wt + ((long)(col0 + 32 + m16)) * Kp + 16 * hi;
  const h16* bp3 = Wt + ((long)(col0 + 48 + m16)) * Kp + 16 * hi;

  v8f acc0 = {}, acc1 = {}, acc2 = {}, acc3 = {};

  for (int k0 = 0; k0 < Kp; k0 += 32) {
    // A fragment: 16x32 f16; lane (m16,hi) holds contiguous K chunk [16hi, 16hi+16)
    v16h a;
    {
      v8h lo = *(const v8h*)(arow + k0 + 16 * hi);
      v8h hv = *(const v8h*)(arow + k0 + 16 * hi + 8);
#pragma unroll
      for (int i = 0; i < 8; ++i) { a[i] = lo[i]; a[i + 8] = hv[i]; }
    }
    __builtin_prefetch((const void*)(arow + k0 + 64), 0, 1);  // next A chunk -> global_prefetch_b8

    // B fragments: 32x16 f16; lane holds contiguous K chunk [16hi, 16hi+16) of its column
    auto loadB = [&](const h16* p) {
      v8h l0 = *(const v8h*)(p + k0);
      v8h l1 = *(const v8h*)(p + k0 + 8);
      v16h b;
#pragma unroll
      for (int i = 0; i < 8; ++i) { b[i] = l0[i]; b[i + 8] = l1[i]; }
      return b;
    };
    acc0 = __builtin_amdgcn_wmma_f32_16x16x32_f16(false, a, false, loadB(bp0), (short)0, acc0, false, false);
    acc1 = __builtin_amdgcn_wmma_f32_16x16x32_f16(false, a, false, loadB(bp1), (short)0, acc1, false, false);
    acc2 = __builtin_amdgcn_wmma_f32_16x16x32_f16(false, a, false, loadB(bp2), (short)0, acc2, false, false);
    acc3 = __builtin_amdgcn_wmma_f32_16x16x32_f16(false, a, false, loadB(bp3), (short)0, acc3, false, false);
  }

  // Epilogue. D layout: lane col = m16, VGPR r -> row (r + 8*hi).
  v8f accs[4] = {acc0, acc1, acc2, acc3};
#pragma unroll
  for (int t = 0; t < 4; ++t) {
    int col = col0 + t * 16 + m16;
    float b = bias[col];
#pragma unroll
    for (int r = 0; r < 8; ++r) {
      long row = row0 + r + 8 * hi;
      float v = accs[t][r] + b;
      if (relu) v = v > 0.f ? v : 0.f;
      C[row * (long)M + col] = (h16)v;
    }
  }
}

// ---------------------------------------------------------------------------
// Segment boundaries: l23 is non-decreasing (both index arrays sorted), so
// batch(i) = l23[i]/Npts is non-decreasing -> binary search per segment.
// ---------------------------------------------------------------------------
__global__ void seg_bounds(const int* __restrict__ l23, int* __restrict__ start,
                           int n3, int npts, int B) {
  int b = threadIdx.x;
  if (b > B) return;
  int target = b * npts, lo = 0, hiq = n3;
  while (lo < hiq) { int mid = (lo + hiq) >> 1; if (l23[mid] < target) lo = mid + 1; else hiq = mid; }
  start[b] = lo;
}

// Segment max over L3X[N3,1024] -> G[128,1024] (rows >= B zero-padded so the
// following 128-row GEMM reads defined data).
__global__ void __launch_bounds__(256) seg_max_pad(const h16* __restrict__ X,
                                                   const int* __restrict__ start,
                                                   h16* __restrict__ G, int C, int B) {
  int b = blockIdx.x;
  int c = blockIdx.y * 256 + threadIdx.x;
  if (c >= C) return;
  if (b >= B) { G[(long)b * C + c] = (h16)0.f; return; }
  int s = start[b], e = start[b + 1];
  float m = -3.0e38f;
  for (int i = s; i < e; ++i) m = fmaxf(m, (float)X[(long)i * C + c]);
  G[(long)b * C + c] = (h16)m;
}

// Head tail: out[i] = softplus( Hb[i,0:64] . w3 + b3 ).  [B,N,1]->[B,1,N] is a
// flat no-op for equal-size graphs.
__global__ void __launch_bounds__(256) head_final(const h16* __restrict__ Hb,
                                                  const float* __restrict__ w3,
                                                  const float* __restrict__ b3,
                                                  float* __restrict__ out, long n) {
  long i = (long)blockIdx.x * 256 + threadIdx.x;
  if (i >= n) return;
  float s = b3[0];
  const h16* h = Hb + i * 64;
#pragma unroll
  for (int k = 0; k < 64; ++k) s += (float)h[k] * w3[k];
  out[i] = s > 0.f ? s + log1pf(expf(-s)) : log1pf(expf(s));
}

// ---------------------------------------------------------------------------
// Host orchestration
// ---------------------------------------------------------------------------
static inline size_t rupb(size_t x) { return (x + 255) & ~(size_t)255; }
static inline int rup32(int k) { return (k + 31) & ~31; }

extern "C" void kernel_launch(void* const* d_in, const int* in_sizes, int n_in,
                              void* d_out, int out_size, void* d_ws, size_t ws_size,
                              hipStream_t stream) {
  (void)in_sizes; (void)n_in; (void)out_size;
  const long N1 = 131072, N2 = 32768, N3 = 8192;
  const int  Bb = 16, Npts = 8192;

  const float* pos  = (const float*)d_in[0];
  const int* l2_idx = (const int*)d_in[40];
  const int* l3_idx = (const int*)d_in[41];
  const int* up21   = (const int*)d_in[42];   // up_l2_from_l3
  const int* up10   = (const int*)d_in[43];   // up_l1_from_l2

  // 18 WMMA GEMM layers (head's final 64->1 handled in head_final).
  struct WSpec { int wi, bi, K, M; };
  static const WSpec sp[18] = {
    {1,2,3,64},    {3,4,64,64},    {5,6,64,128},          // sa1
    {7,8,131,128}, {9,10,128,128}, {11,12,128,256},       // sa2
    {13,14,259,256},{15,16,256,512},{17,18,512,1024},     // sa3
    {19,20,1024,1024},                                     // global
    {21,22,2048,512},{23,24,512,256},                      // fp3
    {25,26,512,256}, {27,28,256,128},                      // fp2
    {29,30,256,128}, {31,32,128,128},                      // fp1
    {33,34,131,128}, {35,36,128,64},                       // head[:2]
  };

  // -------- workspace layout (f16 unless noted); ~175 MB total --------------
  char* ws = (char*)d_ws;
  size_t off = 0;
  auto alloc = [&](size_t bytes) { size_t o = off; off += rupb(bytes); return o; };
  auto Hp = [&](size_t o) { return (h16*)(ws + o); };

  size_t o_L1X = alloc((size_t)N1 * 128 * 2);   // l1_x (later aliased by fp1 out)
  size_t o_L2X = alloc((size_t)N2 * 256 * 2);   // l2_x (later aliased by fp2 out)
  size_t o_L3X = alloc((size_t)N3 * 1024 * 2);  // l3_x (later aliased by fp3 out)
  size_t o_G   = alloc((size_t)128 * 1024 * 2); // pooled, zero-padded to 128 rows
  size_t o_L4X = alloc((size_t)128 * 1024 * 2);
  size_t o_l23 = alloc((size_t)N3 * 4);
  size_t o_seg = alloc(32 * 4);
  size_t o_wt[18];
  for (int i = 0; i < 18; ++i) o_wt[i] = alloc((size_t)sp[i].M * rup32(sp[i].K) * 2);
  size_t o_TA  = alloc((size_t)N1 * 256 * 2);   // ping arena (max: XF1)
  size_t o_TB  = alloc((size_t)N1 * 128 * 2);   // pong arena (max: F1a/Ha)
  if (off > ws_size) return;  // workspace too small: bail deterministically

  int* l23 = (int*)(ws + o_l23);
  int* seg = (int*)(ws + o_seg);

  // -------- weight transforms -----------------------------------------------
  for (int i = 0; i < 18; ++i) {
    int Kp = rup32(sp[i].K);
    dim3 g((unsigned)sp[i].M, (unsigned)((Kp + 255) / 256));
    prep_weight<<<g, 256, 0, stream>>>((const float*)d_in[sp[i].wi], Hp(o_wt[i]),
                                       sp[i].K, sp[i].M, Kp);
  }
  make_l23<<<(unsigned)((N3 + 255) / 256), 256, 0, stream>>>(l3_idx, l2_idx, l23, (int)N3);

  auto gemm = [&](size_t aoff, int s, size_t coff, long rows, int relu) {
    dim3 g((unsigned)(rows / 128), (unsigned)(sp[s].M / 64));
    gemm_bias_act<<<g, 256, 0, stream>>>(Hp(aoff), Hp(o_wt[s]),
                                         (const float*)d_in[sp[s].bi], Hp(coff),
                                         rup32(sp[s].K), sp[s].M, relu);
  };
  auto stage = [&](size_t ooff, int Wp, long rows,
                   const h16* sA, int wA, const int* iA, int dA, int ldA,
                   const h16* sB, int wB, const int* iB, int ldB,
                   const float* sC, int wC, const int* iC, int ldC) {
    dim3 g((unsigned)rows, (unsigned)((Wp + 255) / 256));
    stage_concat<<<g, 256, 0, stream>>>(Hp(ooff), Wp, sA, wA, iA, dA, ldA,
                                        sB, wB, iB, ldB, sC, wC, iC, ldC);
  };

  // -------- sa1: pos[131072,3] -> 64 -> 64 -> 128 ---------------------------
  stage(o_TA, 32, N1, nullptr,0,nullptr,1,0, nullptr,0,nullptr,0, pos,3,nullptr,3);
  gemm(o_TA, 0, o_TB, N1, 1);
  gemm(o_TB, 1, o_TA, N1, 1);
  gemm(o_TA, 2, o_L1X, N1, 0);

  // -------- sa2: [l1_x[l2] | pos[l2]] (131->160) -> 128 -> 128 -> 256 -------
  stage(o_TA, 160, N2, Hp(o_L1X),128,l2_idx,1,128, nullptr,0,nullptr,0, pos,3,l2_idx,3);
  gemm(o_TA, 3, o_TB, N2, 1);
  gemm(o_TB, 4, o_TA, N2, 1);
  gemm(o_TA, 5, o_L2X, N2, 0);

  // -------- sa3: [l2_x[l3] | pos[l23]] (259->288) -> 256 -> 512 -> 1024 -----
  stage(o_TA, 288, N3, Hp(o_L2X),256,l3_idx,1,256, nullptr,0,nullptr,0, pos,3,l23,3);
  gemm(o_TA, 6, o_TB, N3, 1);
  gemm(o_TB, 7, o_TA, N3, 1);
  gemm(o_TA, 8, o_L3X, N3, 0);

  // -------- global max-pool + 1024->1024 ------------------------------------
  seg_bounds<<<1, 32, 0, stream>>>(l23, seg, (int)N3, Npts, Bb);
  { dim3 g(128, 4); seg_max_pad<<<g, 256, 0, stream>>>(Hp(o_L3X), seg, Hp(o_G), 1024, Bb); }
  gemm(o_G, 9, o_L4X, 128, 0);

  // -------- fp3: [l4_x[batch3] | l3_x] (2048) -> 512 -> 256 (aliases L3X) ---
  stage(o_TA, 2048, N3, Hp(o_L4X),1024,l23,Npts,1024, Hp(o_L3X),1024,nullptr,1024,
        nullptr,0,nullptr,0);
  gemm(o_TA, 10, o_TB, N3, 1);
  gemm(o_TB, 11, o_L3X, N3, 0);

  // -------- fp2: [l3_x'[up21] | l2_x] (512) -> 256 -> 128 (aliases L2X) -----
  stage(o_TA, 512, N2, Hp(o_L3X),256,up21,1,256, Hp(o_L2X),256,nullptr,256,
        nullptr,0,nullptr,0);
  gemm(o_TA, 12, o_TB, N2, 1);
  gemm(o_TB, 13, o_L2X, N2, 0);

  // -------- fp1: [l2_x'[up10] | l1_x] (256) -> 128 -> 128 (aliases L1X) -----
  stage(o_TA, 256, N1, Hp(o_L2X),128,up10,1,128, Hp(o_L1X),128,nullptr,128,
        nullptr,0,nullptr,0);
  gemm(o_TA, 14, o_TB, N1, 1);
  gemm(o_TB, 15, o_L1X, N1, 0);

  // -------- head: [l1_x' | pos] (131->160) -> 128(relu) -> 64 -> dot+softplus
  stage(o_TA, 160, N1, Hp(o_L1X),128,nullptr,1,128, nullptr,0,nullptr,0, pos,3,nullptr,3);
  gemm(o_TA, 16, o_TB, N1, 1);
  gemm(o_TB, 17, o_TA, N1, 0);
  head_final<<<(unsigned)((N1 + 255) / 256), 256, 0, stream>>>(
      Hp(o_TA), (const float*)d_in[37], (const float*)d_in[38], (float*)d_out, N1);
}
